// SteinLayer_57775900066258
// MI455X (gfx1250) — matrix operand order repaired and checked by reading, not compiled, not run on hardware
//
#include <hip/hip_runtime.h>
#include <hip/hip_bf16.h>

typedef __attribute__((ext_vector_type(16))) _Float16 v16h;
typedef __attribute__((ext_vector_type(8)))  float    v8f;
typedef __attribute__((ext_vector_type(4)))  float    v4f;

#define DD 32
#define HH 128
#define BLOCK_THREADS 256
#define ROWS_PER_WAVE 32     // two 16-row M-tiles per wave (B tiles reused)
#define ROWS_PER_BLOCK 256   // 8 waves * 32 rows

// Branch-free tanh: prefer the CDNA5 hardware V_TANH_F32.
__device__ __forceinline__ float fast_tanh(float z) {
#if __has_builtin(__builtin_amdgcn_tanhf)
    return __builtin_amdgcn_tanhf(z);
#else
    const float t = __expf(-2.0f * __builtin_fabsf(z));
    const float r = (1.0f - t) * __frcp_rn(1.0f + t);
    return __builtin_copysignf(r, z);
#endif
}

__device__ __forceinline__ v16h load_a_tile(const float* __restrict__ xrow,
                                            int kb0, int kb1,
                                            const float* __restrict__ b2s,
                                            float& xb2) {
    const v4f xa = __builtin_nontemporal_load((const v4f*)(xrow + kb0));
    const v4f xb = __builtin_nontemporal_load((const v4f*)(xrow + kb0 + 4));
    const v4f xc = __builtin_nontemporal_load((const v4f*)(xrow + kb1));
    const v4f xd = __builtin_nontemporal_load((const v4f*)(xrow + kb1 + 4));
    v16h a;
    a[0]  = (_Float16)xa.x; a[1]  = (_Float16)xa.y;
    a[2]  = (_Float16)xa.z; a[3]  = (_Float16)xa.w;
    a[4]  = (_Float16)xb.x; a[5]  = (_Float16)xb.y;
    a[6]  = (_Float16)xb.z; a[7]  = (_Float16)xb.w;
    a[8]  = (_Float16)xc.x; a[9]  = (_Float16)xc.y;
    a[10] = (_Float16)xc.z; a[11] = (_Float16)xc.w;
    a[12] = (_Float16)xd.x; a[13] = (_Float16)xd.y;
    a[14] = (_Float16)xd.z; a[15] = (_Float16)xd.w;
    xb2 = xa.x * b2s[kb0 + 0] + xa.y * b2s[kb0 + 1]
        + xa.z * b2s[kb0 + 2] + xa.w * b2s[kb0 + 3]
        + xb.x * b2s[kb0 + 4] + xb.y * b2s[kb0 + 5]
        + xb.z * b2s[kb0 + 6] + xb.w * b2s[kb0 + 7]
        + xc.x * b2s[kb1 + 0] + xc.y * b2s[kb1 + 1]
        + xc.z * b2s[kb1 + 2] + xc.w * b2s[kb1 + 3]
        + xd.x * b2s[kb1 + 4] + xd.y * b2s[kb1 + 5]
        + xd.z * b2s[kb1 + 6] + xd.w * b2s[kb1 + 7];
    return a;
}

__global__ __launch_bounds__(BLOCK_THREADS)
void stein_wmma_kernel(const float* __restrict__ x,
                       const float* __restrict__ W1,
                       const float* __restrict__ b1,
                       const float* __restrict__ W2,
                       const float* __restrict__ b2,
                       const float* __restrict__ theta0,
                       float* __restrict__ out)
{
    // LDS staging: B-operand layouts, f16.
    // w1t[j][k] = W1[k][j]  (B1 = W1, 32x128 -> per-column contiguous K)
    // w2h[j][k] = W2[j][k]  (B2 = W2^T, so [col=j][k=d] == W2 row-major)
    __shared__ __attribute__((aligned(32))) _Float16 w1t[HH * DD];
    __shared__ __attribute__((aligned(32))) _Float16 w2h[HH * DD];
    __shared__ float b1s[HH];
    __shared__ float b2s[DD];
    __shared__ float cs[HH];     // c_j = sum_d W1[d,j] * W2[j,d]
    __shared__ float csum_s;

    const int tid = threadIdx.x;

    for (int idx = tid; idx < HH * DD; idx += BLOCK_THREADS) {
        const int j = idx & (HH - 1);   // fastest -> coalesced W1 read
        const int k = idx >> 7;
        w1t[j * DD + k] = (_Float16)W1[k * HH + j];
        w2h[idx]        = (_Float16)W2[idx];
    }
    if (tid < HH) {
        b1s[tid] = b1[tid];
        float s = 0.f;
        #pragma unroll
        for (int d = 0; d < DD; ++d)
            s += W1[d * HH + tid] * W2[tid * DD + d];
        cs[tid] = s;
    }
    if (tid < DD) b2s[tid] = b2[tid];
    __syncthreads();
    if (tid == 0) {
        float s = 0.f;
        for (int j = 0; j < HH; ++j) s += cs[j];
        csum_s = s;
    }
    __syncthreads();

    const int wave = tid >> 5;
    const int lane = tid & 31;
    const int row  = lane & 15;      // M index within tile (A) / N col (B,C)
    const int half = lane >> 4;
    const int m0   = blockIdx.x * ROWS_PER_BLOCK + wave * ROWS_PER_WAVE;

    // ---- Build two A tiles (16x32 f16 each): lane holds row `row`, K in
    // {half*16 .. +7} and {half*16+8 .. +7} (contiguous, matches B).
    const int kb0 = half * 16;
    const int kb1 = half * 16 + 8;
    float xb2_0, xb2_1;
    const v16h a0 = load_a_tile(x + (size_t)(m0 + row) * DD,      kb0, kb1, b2s, xb2_0);
    const v16h a1 = load_a_tile(x + (size_t)(m0 + 16 + row) * DD, kb0, kb1, b2s, xb2_1);
    // Partner lane (lane^16) holds the complementary K set of the same row.
    xb2_0 += __shfl_xor(xb2_0, 16, 32);
    xb2_1 += __shfl_xor(xb2_1, 16, 32);

    v8f racc0 = {0.f, 0.f, 0.f, 0.f, 0.f, 0.f, 0.f, 0.f};
    v8f racc1 = {0.f, 0.f, 0.f, 0.f, 0.f, 0.f, 0.f, 0.f};
    const v8f zerov = {};

    // 8 N-tiles of 16 hidden units each: Z = X*W1 + b1 ; G = X*W2^T.
    // B tiles are loaded once from LDS and reused for both M-tiles.
    #pragma unroll
    for (int n = 0; n < 8; ++n) {
        const int col = n * 16 + row;             // this lane's hidden index j
        const float b1j = b1s[col];
        const float cj  = cs[col];

        // B operands: lane col = `row`, K = half*16 .. half*16+15 contiguous
        const v16h bw1 = *(const v16h*)(&w1t[col * DD + half * 16]);
        const v16h bw2 = *(const v16h*)(&w2h[col * DD + half * 16]);

        const v8f cin = {b1j, b1j, b1j, b1j, b1j, b1j, b1j, b1j};

        v8f z0 = __builtin_amdgcn_wmma_f32_16x16x32_f16(
                     false, a0, false, bw1, (short)0, cin, false, false);
        v8f z1 = __builtin_amdgcn_wmma_f32_16x16x32_f16(
                     false, a1, false, bw1, (short)0, cin, false, false);
        v8f g0 = __builtin_amdgcn_wmma_f32_16x16x32_f16(
                     false, a0, false, bw2, (short)0, zerov, false, false);
        v8f g1 = __builtin_amdgcn_wmma_f32_16x16x32_f16(
                     false, a1, false, bw2, (short)0, zerov, false, false);

        #pragma unroll
        for (int r = 0; r < 8; ++r) {
            const float h0 = fast_tanh(z0[r]);
            const float h1 = fast_tanh(z1[r]);
            // accumulate  -h*g - h^2*c_j  = -(h*(g + h*c_j))
            racc0[r] -= h0 * (g0[r] + h0 * cj);
            racc1[r] -= h1 * (g1[r] + h1 * cj);
        }
    }

    // Reduce each row accumulator across the 16 lanes of its half
    // (C layout: VGPR r holds row r for lanes 0-15, row r+8 for lanes 16-31)
    #pragma unroll
    for (int r = 0; r < 8; ++r) {
        float v0 = racc0[r];
        float v1 = racc1[r];
        v0 += __shfl_xor(v0, 1, 32);
        v1 += __shfl_xor(v1, 1, 32);
        v0 += __shfl_xor(v0, 2, 32);
        v1 += __shfl_xor(v1, 2, 32);
        v0 += __shfl_xor(v0, 4, 32);
        v1 += __shfl_xor(v1, 4, 32);
        v0 += __shfl_xor(v0, 8, 32);
        v1 += __shfl_xor(v1, 8, 32);
        racc0[r] = v0;
        racc1[r] = v1;
    }

    const float th = theta0[0];
    // Writers: lanes 0..7 -> rows 0..7 (half 0), lanes 24..31 -> rows 8..15
    // (half 1).  For both, written row == (lane & 15) so xb2 matches.
    if (lane < 8 || lane >= 24) {
        const int r = lane & 7;
        float v0 = 0.f, v1 = 0.f;
        #pragma unroll
        for (int q = 0; q < 8; ++q)
            if (q == r) { v0 = racc0[q]; v1 = racc1[q]; }
        const float base = csum_s + th;
        out[m0 + (lane & 15)]      = v0 + base - xb2_0;
        out[m0 + 16 + (lane & 15)] = v1 + base - xb2_1;
    }
}

extern "C" void kernel_launch(void* const* d_in, const int* in_sizes, int n_in,
                              void* d_out, int out_size, void* d_ws, size_t ws_size,
                              hipStream_t stream) {
    const float* x   = (const float*)d_in[0];
    const float* W1  = (const float*)d_in[1];
    const float* b1  = (const float*)d_in[2];
    const float* W2  = (const float*)d_in[3];
    const float* b2  = (const float*)d_in[4];
    const float* th  = (const float*)d_in[5];
    float* out = (float*)d_out;

    const int Bn = in_sizes[0] / DD;           // 65536
    const int grid = Bn / ROWS_PER_BLOCK;      // 256 blocks of 8 waves
    stein_wmma_kernel<<<grid, BLOCK_THREADS, 0, stream>>>(x, W1, b1, W2, b2, th, out);
}
